// EmbeddingLSTM_55044300866145
// MI455X (gfx1250) — matrix operand order repaired and checked by reading, not compiled, not run on hardware
//
#include <hip/hip_runtime.h>
#include <hip/hip_bf16.h>
#include <math.h>

typedef __attribute__((ext_vector_type(16))) _Float16 v16h;
typedef __attribute__((ext_vector_type(8)))  float    v8f;

// Problem constants
#define BB 64
#define SS 128
#define TT 256
#define HH 1024
#define KK 1280           // IN + H = 256 + 1024
#define KT 40             // K tiles of 32
#define NHT 64            // H/16 hidden-column tiles

// ---------------------------------------------------------------------------
// Zero the recurrent state (h in f16, c in f32) at the start of every launch.
// ---------------------------------------------------------------------------
__global__ void lstm_init_state(_Float16* __restrict__ hbuf, float* __restrict__ cbuf) {
    int i = blockIdx.x * 256 + threadIdx.x;
    if (i < BB * HH) {
        hbuf[i] = (_Float16)0.f;
        cbuf[i] = 0.f;
    }
}

// ---------------------------------------------------------------------------
// Pack the 4 gate weight matrices [1280,1024] (fp32, row-major) into f16
// WMMA B-fragments.  Layout: [nh(64)][gate(4)][kt(40)][lane(32)][elem(16)].
// B-frag (16x16x32 f16): lane<16 -> N=lane, K=kt*32+j ; lane>=16 -> N=lane-16,
// K=kt*32+16+j.
// ---------------------------------------------------------------------------
__global__ void lstm_pack_B(const float* __restrict__ Wf, const float* __restrict__ Wi,
                            const float* __restrict__ Wu, const float* __restrict__ Wo,
                            _Float16* __restrict__ pB) {
    int idx  = blockIdx.x * 256 + threadIdx.x;      // < 64*4*40*32 = 327680
    int lane = idx & 31;
    int rest = idx >> 5;
    int kt   = rest % KT;  rest /= KT;
    int gate = rest & 3;
    int nh   = rest >> 2;
    const float* W = (gate == 0) ? Wf : (gate == 1) ? Wi : (gate == 2) ? Wu : Wo;
    int n     = nh * 16 + (lane & 15);
    int kbase = kt * 32 + ((lane >= 16) ? 16 : 0);
    _Float16* dst = pB + (size_t)idx * 16;
#pragma unroll
    for (int j = 0; j < 16; ++j)
        dst[j] = (_Float16)W[(size_t)(kbase + j) * HH + n];
}

// ---------------------------------------------------------------------------
// Per-step: pack z = [interleave(x_t, y_{t-1}) | h_prev]  (64 x 1280) into f16
// WMMA A-fragments.  Layout: [mt(4)][kt(40)][lane(32)][elem(16)].
// A-frag (16x16x32 f16): lane<16 -> M=lane,   K = kt*32 + j
//                        lane>=16-> M=lane-16,K = kt*32 + 16 + j
// ---------------------------------------------------------------------------
__global__ void lstm_pack_A(const float* __restrict__ x, const float* __restrict__ y,
                            const _Float16* __restrict__ hbuf, _Float16* __restrict__ pA,
                            int t) {
    int idx  = blockIdx.x * 256 + threadIdx.x;      // < 4*40*32 = 5120
    int lane = idx & 31;
    int rest = idx >> 5;
    int kt   = rest % KT;
    int mt   = rest / KT;
    int b    = mt * 16 + (lane & 15);
    int kadd = (lane >= 16) ? 16 : 0;
    _Float16* dst = pA + (size_t)idx * 16;
#pragma unroll
    for (int j = 0; j < 16; ++j) {
        int k = kt * 32 + j + kadd;
        float v;
        if (k < 2 * SS) {                       // interleaved (x_t, y_{t-1})
            int s = k >> 1;
            if ((k & 1) == 0) v = x[((size_t)b * SS + s) * TT + t];
            else              v = (t > 0) ? y[((size_t)b * SS + s) * TT + (t - 1)] : 0.f;
        } else {                                // h_prev
            v = (float)hbuf[(size_t)b * HH + (k - 2 * SS)];
        }
        dst[j] = (_Float16)v;
    }
}

// ---------------------------------------------------------------------------
// One LSTM step.  Block = 256 threads (8 waves), grid = 64 hidden-col tiles.
// Wave w: m-tile = w&3, gate pair = w>>2 ({f,i} or {u,o}); 2 WMMA chains per
// wave reuse the A fragment.  Gates exchanged through LDS, then elementwise
// c/h update + output write.
// ---------------------------------------------------------------------------
__global__ __launch_bounds__(256) void lstm_step(
        const _Float16* __restrict__ pA, const _Float16* __restrict__ pB,
        const float* __restrict__ bf, const float* __restrict__ bi,
        const float* __restrict__ bu, const float* __restrict__ bo,
        float* __restrict__ cbuf, _Float16* __restrict__ hbuf,
        float* __restrict__ out, float* __restrict__ cT, int t) {
    __shared__ float g_s[4][64][17];            // [gate][batch row][col], padded

    const int nh   = blockIdx.x;
    const int tid  = threadIdx.x;
    const int wave = tid >> 5;
    const int lane = tid & 31;
    const int mt   = wave & 3;
    const int gp   = wave >> 2;
    const int g0   = gp * 2;
    const int g1   = gp * 2 + 1;

    const v16h* aBase  = (const v16h*)pA + (size_t)(mt * KT) * 32 + lane;
    const v16h* b0Base = (const v16h*)pB + (size_t)((nh * 4 + g0) * KT) * 32 + lane;
    const v16h* b1Base = (const v16h*)pB + (size_t)((nh * 4 + g1) * KT) * 32 + lane;

    v8f acc0 = {};
    v8f acc1 = {};
    for (int kt = 0; kt < KT; ++kt) {
        v16h a  = aBase [kt * 32];
        v16h w0 = b0Base[kt * 32];
        v16h w1 = b1Base[kt * 32];
        acc0 = __builtin_amdgcn_wmma_f32_16x16x32_f16(false, a, false, w0,
                                                      (short)0, acc0, false, false);
        acc1 = __builtin_amdgcn_wmma_f32_16x16x32_f16(false, a, false, w1,
                                                      (short)0, acc1, false, false);
    }

    // C/D layout: lane<16 -> rows v..v (M=v), lane>=16 -> M=8+v; N = lane%16
    const int rbase = mt * 16 + ((lane >= 16) ? 8 : 0);
    const int col   = lane & 15;
#pragma unroll
    for (int v = 0; v < 8; ++v) {
        g_s[g0][rbase + v][col] = acc0[v];
        g_s[g1][rbase + v][col] = acc1[v];
    }
    __syncthreads();

    // Elementwise LSTM update: 64 rows x 16 cols = 1024 cells, 4 per thread.
#pragma unroll
    for (int i = 0; i < 4; ++i) {
        int e  = tid + 256 * i;
        int b  = e >> 4;
        int j  = e & 15;
        int hc = nh * 16 + j;
        float fg = g_s[0][b][j] + bf[hc];
        float ig = g_s[1][b][j] + bi[hc];
        float ug = g_s[2][b][j] + bu[hc];
        float og = g_s[3][b][j] + bo[hc];
        float sf = 1.f / (1.f + __expf(-fg));
        float si = 1.f / (1.f + __expf(-ig));
        float so = 1.f / (1.f + __expf(-og));
        size_t sidx = (size_t)b * HH + hc;
        float cn = cbuf[sidx] * sf + si * tanhf(ug);
        float h  = so * tanhf(cn);
        cbuf[sidx] = cn;
        hbuf[sidx] = (_Float16)h;
        out[sidx * TT + t] = h;                 // out[B][H][T]
        if (t == TT - 1) cT[sidx] = cn;         // final cell state
    }
}

// ---------------------------------------------------------------------------
// Host-side orchestration (all on `stream`, graph-capture safe).
// ---------------------------------------------------------------------------
extern "C" void kernel_launch(void* const* d_in, const int* in_sizes, int n_in,
                              void* d_out, int out_size, void* d_ws, size_t ws_size,
                              hipStream_t stream) {
    const float* x  = (const float*)d_in[0];
    const float* y  = (const float*)d_in[2];
    const float* Wf = (const float*)d_in[3];
    const float* bf = (const float*)d_in[4];
    const float* Wi = (const float*)d_in[5];
    const float* bi = (const float*)d_in[6];
    const float* Wu = (const float*)d_in[7];
    const float* bu = (const float*)d_in[8];
    const float* Wo = (const float*)d_in[9];
    const float* bo = (const float*)d_in[10];

    // Workspace layout (all 256B-aligned sections)
    char* ws = (char*)d_ws;
    const size_t szB = (size_t)NHT * 4 * KT * 32 * 16 * sizeof(_Float16); // 10,485,760
    const size_t szA = (size_t)4 * KT * 32 * 16 * sizeof(_Float16);       //    163,840
    const size_t szH = (size_t)BB * HH * sizeof(_Float16);                //    131,072
    _Float16* packedB = (_Float16*)ws;
    _Float16* packedA = (_Float16*)(ws + szB);
    _Float16* hbuf    = (_Float16*)(ws + szB + szA);
    float*    cbuf    = (float*)   (ws + szB + szA + szH);

    float* out = (float*)d_out;
    float* cT  = out + (size_t)BB * HH * TT;

    lstm_init_state<<<(BB * HH + 255) / 256, 256, 0, stream>>>(hbuf, cbuf);
    lstm_pack_B<<<(NHT * 4 * KT * 32) / 256, 256, 0, stream>>>(Wf, Wi, Wu, Wo, packedB);

    for (int t = 0; t < TT; ++t) {
        lstm_pack_A<<<(4 * KT * 32) / 256, 256, 0, stream>>>(x, y, hbuf, packedA, t);
        lstm_step<<<NHT, 256, 0, stream>>>(packedA, packedB, bf, bi, bu, bo,
                                           cbuf, hbuf, out, cT, t);
    }
}
